// MultiHeadAttention_63866163691555
// MI455X (gfx1250) — matrix operand order repaired and checked by reading, not compiled, not run on hardware
//
#include <hip/hip_runtime.h>
#include <hip/hip_bf16.h>

#define D_MODEL 1024
#define NHEAD   16
#define D_K     64

typedef __attribute__((ext_vector_type(16))) _Float16 v16h;
typedef __attribute__((ext_vector_type(8)))  float    v8f;
typedef __attribute__((address_space(3)))    char     lds_c;
typedef __attribute__((address_space(3)))    _Float16 lds_h;
typedef __attribute__((address_space(3)))    float    lds_f;

// ---------------------------------------------------------------------------
// WMMA helper: D = A(16x32 f16) x B(32x16 f16) + C(16x16 f32)
// ---------------------------------------------------------------------------
__device__ __forceinline__ v8f wmma16(v16h a, v16h b, v8f c) {
  return __builtin_amdgcn_wmma_f32_16x16x32_f16(
      /*neg_a=*/false, a, /*neg_b=*/false, b,
      /*c_mod=*/(short)0, c, /*reuse_a=*/false, /*reuse_b=*/false);
}

// ---------------------------------------------------------------------------
// Fragment loaders (templated over pointer type: generic/global or AS(3) LDS
// so LDS accesses are guaranteed to lower to ds_load_b128, never flat).
// ISA layouts (CDNA5 ISA §7.12.2):
//  A 16x32 (MxK): lane m   (0-15) : K k0+0..7   (elems 0-7), K k0+16..23 (elems 8-15)
//                 lane m+16       : K k0+8..15  (elems 0-7), K k0+24..31 (elems 8-15)
//  B 32x16 (KxN): lane n   (0-15) : K k0+0..7   (elems 0-7), K k0+16..23 (elems 8-15)
//                 lane n+16       : K k0+8..15  (elems 0-7), K k0+24..31 (elems 8-15)
//  C 16x16:       vgpr j: lanes 0-15 -> (row j, col lane), lanes 16-31 -> (row j+8)
// ---------------------------------------------------------------------------
template <typename P>
__device__ __forceinline__ v16h loadA_T(P base, long ld, int row0, int k0) {
  const int lane = threadIdx.x & 31;
  P p = base + (long)(row0 + (lane & 15)) * ld + (k0 + ((lane >> 4) << 3));
  v16h r;
#pragma unroll
  for (int i = 0; i < 8; ++i) { r[i] = p[i]; r[8 + i] = p[16 + i]; }
  return r;
}

template <typename P>
__device__ __forceinline__ v16h loadB_T(P base, long ld, int n0, int k0) {
  const int lane = threadIdx.x & 31;
  P p = base + (long)(n0 + (lane & 15)) * ld + (k0 + ((lane >> 4) << 3));
  v16h r;
#pragma unroll
  for (int i = 0; i < 8; ++i) { r[i] = p[i]; r[8 + i] = p[16 + i]; }
  return r;
}

// ---------------------------------------------------------------------------
// CDNA5 async copy: global -> LDS, 16B per lane, tracked by ASYNCcnt.
// ---------------------------------------------------------------------------
__device__ __forceinline__ void async_b128(lds_h* lds_ptr, const _Float16* gptr) {
  unsigned lo = (unsigned)(size_t)lds_ptr;                  // 32-bit LDS address
  unsigned long long ga = (unsigned long long)(size_t)gptr; // 64-bit global address
  asm volatile("global_load_async_to_lds_b128 %0, %1, off"
               :: "v"(lo), "v"(ga) : "memory");
}
__device__ __forceinline__ void wait_async0() {
  asm volatile("s_wait_asynccnt 0" ::: "memory");
}
__device__ __forceinline__ void wait_ds0() {
  asm volatile("s_wait_dscnt 0" ::: "memory");
}

// ---------------------------------------------------------------------------
// fp32 -> f16 conversion (each input read exactly once).
// ---------------------------------------------------------------------------
__global__ __launch_bounds__(256)
void cvt_f16_kernel(const float* __restrict__ in, _Float16* __restrict__ out) {
  const long i = ((long)blockIdx.x * 256 + threadIdx.x) * 8;
#pragma unroll
  for (int j = 0; j < 8; ++j) out[i + j] = (_Float16)in[i + j];
}

// ---------------------------------------------------------------------------
// GEMM: C[M,N] = A[M,K](f16) @ W[N,K](f16)^T, f32 accumulate.
// Block tile 128x64 (8 waves, each a 32x32 = 2x2 WMMA tile), K-step 32.
// Double-buffered LDS tiles filled by global_load_async_to_lds_b128.
// Epilogue bounces C through LDS for coalesced 64/128B-per-thread stores.
// ---------------------------------------------------------------------------
#define LDT 40  // padded LDS row stride in halves (32 data + 8 pad)

__global__ __launch_bounds__(256)
void gemm_awt_kernel(const _Float16* __restrict__ A, const _Float16* __restrict__ W,
                     int M, int N, int K,
                     _Float16* __restrict__ outH, float* __restrict__ outF,
                     const float* __restrict__ bias)
{
  // 32KB: A/B double buffers (30720B) during mainloop, C staging in epilogue.
  __shared__ __align__(16) char smem[32768];
  lds_h* const Sh = (lds_h*)(lds_c*)smem;     // AS(3) halves view
  auto Abuf = [&](int b) -> lds_h* { return Sh + b * (128 * LDT); };
  auto Bbuf = [&](int b) -> lds_h* { return Sh + 2 * 128 * LDT + b * (64 * LDT); };

  const int tid  = threadIdx.x;
  const int lane = tid & 31;
  const int wave = tid >> 5;
  const int rowB = blockIdx.y * 128;
  const int colB = blockIdx.x * 64;
  const int row0 = (wave & 3) * 32;   // wave tile inside block
  const int col0 = (wave >> 2) * 32;

  // --- async stage of one K-tile (A: 512 16B chunks, B: 256 16B chunks) ----
  auto stage = [&](int buf, int k0) {
#pragma unroll
    for (int q = 0; q < 2; ++q) {          // A: 2 chunks / thread
      const int c = tid * 2 + q;
      const int r = c >> 2, ch = (c & 3) * 8;
      async_b128(Abuf(buf) + r * LDT + ch, A + (long)(rowB + r) * K + k0 + ch);
    }
    {                                      // B: 1 chunk / thread
      const int r = tid >> 2, ch = (tid & 3) * 8;
      async_b128(Bbuf(buf) + r * LDT + ch, W + (long)(colB + r) * K + k0 + ch);
    }
  };

  stage(0, 0);
  wait_async0();
  __syncthreads();

  const v8f vzero = {0.f, 0.f, 0.f, 0.f, 0.f, 0.f, 0.f, 0.f};
  v8f acc[2][2];
#pragma unroll
  for (int i = 0; i < 2; ++i)
#pragma unroll
    for (int j = 0; j < 2; ++j) acc[i][j] = vzero;

  int buf = 0;
  for (int k0 = 0; k0 < K; k0 += 32) {
    const bool more = (k0 + 32 < K);
    if (more) stage(buf ^ 1, k0 + 32);     // overlap DMA with WMMA

    v16h a0 = loadA_T<const lds_h*>(Abuf(buf), LDT, row0,      0);
    v16h a1 = loadA_T<const lds_h*>(Abuf(buf), LDT, row0 + 16, 0);
    v16h b0 = loadB_T<const lds_h*>(Bbuf(buf), LDT, col0,      0);
    v16h b1 = loadB_T<const lds_h*>(Bbuf(buf), LDT, col0 + 16, 0);
    acc[0][0] = wmma16(a0, b0, acc[0][0]);
    acc[0][1] = wmma16(a0, b1, acc[0][1]);
    acc[1][0] = wmma16(a1, b0, acc[1][0]);
    acc[1][1] = wmma16(a1, b1, acc[1][1]);

    if (more) {
      wait_async0();      // next tile resident
      __syncthreads();    // all waves done reading current tile
    }
    buf ^= 1;
  }

  // ---- epilogue: stage C in LDS, then coalesced stores ---------------------
  __syncthreads();        // done with A/B tiles; smem becomes C staging
  const int nn = lane & 15;
  const int mb = (lane >> 4) << 3;
  if (outH) {
    lds_h* Cs = Sh;                   // [128][64] f16 (16KB)
#pragma unroll
    for (int i = 0; i < 2; ++i)
#pragma unroll
      for (int j = 0; j < 2; ++j) {
        const int cc = col0 + j * 16 + nn;
        const int rr = row0 + i * 16 + mb;
#pragma unroll
        for (int e = 0; e < 8; ++e)
          Cs[(rr + e) * 64 + cc] = (_Float16)acc[i][j][e];
      }
    __syncthreads();
    const int r  = tid >> 1;              // 0..127
    const int c0 = (tid & 1) * 32;        // 64B contiguous run per thread
    _Float16*    gp = outH + (long)(rowB + r) * N + colB + c0;
    const lds_h* lp = Cs + r * 64 + c0;
#pragma unroll
    for (int i = 0; i < 32; ++i) gp[i] = lp[i];
  } else {
    lds_f* Cs = (lds_f*)(lds_c*)smem;  // [128][64] f32 (32KB)
#pragma unroll
    for (int i = 0; i < 2; ++i)
#pragma unroll
      for (int j = 0; j < 2; ++j) {
        const int cc = col0 + j * 16 + nn;
        const int rr = row0 + i * 16 + mb;
#pragma unroll
        for (int e = 0; e < 8; ++e)
          Cs[(rr + e) * 64 + cc] = acc[i][j][e];
      }
    __syncthreads();
    const int r  = tid >> 1;
    const int c0 = (tid & 1) * 32;        // 128B contiguous run per thread
    float*       gp = outF + (long)(rowB + r) * N + colB + c0;
    const lds_f* lp = Cs + r * 64 + c0;
    if (bias) {
      const float* bp = bias + colB + c0;
#pragma unroll
      for (int i = 0; i < 32; ++i) gp[i] = lp[i] + bp[i];
    } else {
#pragma unroll
      for (int i = 0; i < 32; ++i) gp[i] = lp[i];
    }
  }
}

// ---------------------------------------------------------------------------
// Causal flash attention. One workgroup = (b, h, 128 q rows), 8 waves,
// each wave owns 16 q rows. KV processed in blocks of 64.
// q/k/v stored f16 as [B*S, 1024] = [b, s, h*64 + d]; y written f16.
// ---------------------------------------------------------------------------
__global__ __launch_bounds__(256)
void flash_attn_kernel(const _Float16* __restrict__ Qp,
                       const _Float16* __restrict__ Kp,
                       const _Float16* __restrict__ Vp,
                       _Float16* __restrict__ Y, int S)
{
  __shared__ _Float16 Plds[8][16 * 64];   // per-wave P tile, row-major [16][64]
  __shared__ _Float16 Vts[64][72];        // V tile transposed: [d][kv], padded

  const int lane = threadIdx.x & 31;
  const int wave = threadIdx.x >> 5;
  const int qb   = blockIdx.x * 128;
  const int h    = blockIdx.y;
  const int b    = blockIdx.z;
  const int qr0  = qb + wave * 16;
  const long ld  = D_MODEL;

  lds_h* const pb = (lds_h*)&Plds[wave][0];  // AS(3) views -> guaranteed ds ops
  lds_h* const vt = (lds_h*)&Vts[0][0];

  const _Float16* Qb = Qp + (long)b * S * ld + h * D_K;
  const _Float16* Kb = Kp + (long)b * S * ld + h * D_K;
  const _Float16* Vb = Vp + (long)b * S * ld + h * D_K;
  _Float16*       Yb = Y  + (long)b * S * ld + h * D_K;

  // q fragments for this wave's 16 rows (K-dim = 64 -> two 16x32 A frags)
  const v16h qa0 = loadA_T<const _Float16*>(Qb, ld, qr0, 0);
  const v16h qa1 = loadA_T<const _Float16*>(Qb, ld, qr0, 32);

  const v8f vzero = {0.f, 0.f, 0.f, 0.f, 0.f, 0.f, 0.f, 0.f};
  v8f o[4];
#pragma unroll
  for (int t = 0; t < 4; ++t) o[t] = vzero;
  float mrow[8], lrow[8];
#pragma unroll
  for (int j = 0; j < 8; ++j) { mrow[j] = -1e30f; lrow[j] = 0.f; }

  const int nn = lane & 15;
  const int mb = (lane >> 4) << 3;
  const float scale = 0.125f;  // 1/sqrt(64)

  for (int kv0 = 0; kv0 < qb + 128; kv0 += 64) {
    __syncthreads();  // protect Vt from previous iteration's readers
    {   // cooperatively stage V[kv0..kv0+63][0..63] transposed into Vt[d][kv]
      const int kv = threadIdx.x >> 2;
      const int d0 = (threadIdx.x & 3) * 16;
      const _Float16* vp = Vb + (long)(kv0 + kv) * ld + d0;
#pragma unroll
      for (int i = 0; i < 16; ++i) vt[(d0 + i) * 72 + kv] = vp[i];
      if (kv0 + 64 < qb + 128) {  // prefetch next KV block of K and V
        __builtin_prefetch(Kb + (long)(kv0 + 64 + kv) * ld, 0, 1);
        __builtin_prefetch(Vb + (long)(kv0 + 64 + kv) * ld, 0, 1);
      }
    }
    __syncthreads();

    if (kv0 <= qr0 + 15) {  // wave-uniform: this wave has unmasked columns
      // ---- scores S[16,64] = q @ k^T  (k rows are d-contiguous => B frags) --
      v8f s[4];
#pragma unroll
      for (int t = 0; t < 4; ++t) {
        s[t] = vzero;
        v16h kb0 = loadB_T<const _Float16*>(Kb, ld, kv0 + t * 16, 0);
        v16h kb1 = loadB_T<const _Float16*>(Kb, ld, kv0 + t * 16, 32);
        s[t] = wmma16(qa0, kb0, s[t]);
        s[t] = wmma16(qa1, kb1, s[t]);
      }
      // ---- online softmax (row = qr0+mb+j, col = kv0+t*16+nn) --------------
#pragma unroll
      for (int j = 0; j < 8; ++j) {
        const int row = qr0 + mb + j;
        float mx = -1e30f;
#pragma unroll
        for (int t = 0; t < 4; ++t) {
          float sv = s[t][j] * scale;
          if (kv0 + t * 16 + nn > row) sv = -1e30f;  // causal mask
          s[t][j] = sv;
          mx = fmaxf(mx, sv);
        }
        for (int msk = 1; msk <= 8; msk <<= 1)      // 16-lane butterfly
          mx = fmaxf(mx, __shfl_xor(mx, msk, 32));
        const float mn    = fmaxf(mrow[j], mx);
        const float alpha = __expf(mrow[j] - mn);
        float rs = 0.f;
#pragma unroll
        for (int t = 0; t < 4; ++t) {
          const float p = __expf(s[t][j] - mn);
          s[t][j] = p;
          rs += p;
        }
        for (int msk = 1; msk <= 8; msk <<= 1)
          rs += __shfl_xor(rs, msk, 32);
        lrow[j] = lrow[j] * alpha + rs;
        mrow[j] = mn;
#pragma unroll
        for (int t = 0; t < 4; ++t) o[t][j] *= alpha;
      }
      // ---- stage P (C layout -> row-major f16 in LDS) ----------------------
#pragma unroll
      for (int t = 0; t < 4; ++t)
#pragma unroll
        for (int j = 0; j < 8; ++j)
          pb[(mb + j) * 64 + t * 16 + nn] = (_Float16)s[t][j];
      wait_ds0();  // wave-local DS RAW
      // ---- O += P[16,64] @ V[64,64]  (B frags from transposed Vt) ----------
#pragma unroll
      for (int kk = 0; kk < 2; ++kk) {
        v16h pa = loadA_T<const lds_h*>(pb, 64, 0, kk * 32);
#pragma unroll
        for (int t = 0; t < 4; ++t) {
          v16h vb = loadB_T<const lds_h*>(vt, 72, t * 16, kk * 32);
          o[t] = wmma16(pa, vb, o[t]);
        }
      }
    }
  }

  // ---- normalize, stage O in LDS, coalesced y store (f16) ------------------
#pragma unroll
  for (int t = 0; t < 4; ++t)
#pragma unroll
    for (int j = 0; j < 8; ++j)
      pb[(mb + j) * 64 + t * 16 + nn] = (_Float16)(o[t][j] / lrow[j]);
  wait_ds0();  // wave-local DS RAW
  {
    const int r  = lane >> 1;           // 0..15
    const int c0 = (lane & 1) * 32;     // 64B contiguous run per lane
    _Float16*    gp = Yb + (long)(qr0 + r) * ld + c0;
    const lds_h* lp = pb + r * 64 + c0;
#pragma unroll
    for (int i = 0; i < 32; ++i) gp[i] = lp[i];
  }
}

// ---------------------------------------------------------------------------
extern "C" void kernel_launch(void* const* d_in, const int* in_sizes, int n_in,
                              void* d_out, int out_size, void* d_ws, size_t ws_size,
                              hipStream_t stream) {
  const float* Q   = (const float*)d_in[0];
  const float* K   = (const float*)d_in[1];
  const float* V   = (const float*)d_in[2];
  const float* WQ  = (const float*)d_in[3];
  const float* WK  = (const float*)d_in[4];
  const float* WV  = (const float*)d_in[5];
  const float* Wfc = (const float*)d_in[6];
  const float* bfc = (const float*)d_in[7];
  float* out = (float*)d_out;

  const int Dm = D_MODEL;
  const int M  = in_sizes[0] / Dm;   // B*S = 4096
  const int B  = 2;
  const int S  = M / B;              // 2048

  // Workspace layout (halves): Qh|Kh|Vh (M*Dm) | WQh|WKh|WVh|Wfch (Dm*Dm)
  //                            | qh|kh|vh (M*Dm) | yh (M*Dm)
  const size_t MD = (size_t)M * Dm, DD = (size_t)Dm * Dm;
  _Float16* Qh   = (_Float16*)d_ws;
  _Float16* Kh   = Qh + MD;
  _Float16* Vh   = Kh + MD;
  _Float16* WQh  = Vh + MD;
  _Float16* WKh  = WQh + DD;
  _Float16* WVh  = WKh + DD;
  _Float16* Wfch = WVh + DD;
  _Float16* qh   = Wfch + DD;
  _Float16* kh   = qh + MD;
  _Float16* vh   = kh + MD;
  _Float16* yh   = vh + MD;

  const dim3 bb(256);
  // one-time fp32 -> f16 conversions (each input read once)
  cvt_f16_kernel<<<dim3(MD / (8 * 256)), bb, 0, stream>>>(Q,  Qh);
  cvt_f16_kernel<<<dim3(MD / (8 * 256)), bb, 0, stream>>>(K,  Kh);
  cvt_f16_kernel<<<dim3(MD / (8 * 256)), bb, 0, stream>>>(V,  Vh);
  cvt_f16_kernel<<<dim3(DD / (8 * 256)), bb, 0, stream>>>(WQ,  WQh);
  cvt_f16_kernel<<<dim3(DD / (8 * 256)), bb, 0, stream>>>(WK,  WKh);
  cvt_f16_kernel<<<dim3(DD / (8 * 256)), bb, 0, stream>>>(WV,  WVh);
  cvt_f16_kernel<<<dim3(DD / (8 * 256)), bb, 0, stream>>>(Wfc, Wfch);

  const dim3 gg(Dm / 64, M / 128);
  gemm_awt_kernel<<<gg, bb, 0, stream>>>(Qh, WQh, M, Dm, Dm, qh, nullptr, nullptr);
  gemm_awt_kernel<<<gg, bb, 0, stream>>>(Kh, WKh, M, Dm, Dm, kh, nullptr, nullptr);
  gemm_awt_kernel<<<gg, bb, 0, stream>>>(Vh, WVh, M, Dm, Dm, vh, nullptr, nullptr);

  flash_attn_kernel<<<dim3(S / 128, NHEAD, B), bb, 0, stream>>>(qh, kh, vh, yh, S);

  gemm_awt_kernel<<<gg, bb, 0, stream>>>(yh, Wfch, M, Dm, Dm, nullptr, out, bfc);
}
